// ContrastiveLoss_76819785056610
// MI455X (gfx1250) — matrix operand order repaired and checked by reading, not compiled, not run on hardware
//
#include <hip/hip_runtime.h>
#include <hip/hip_bf16.h>

// ---------------------------------------------------------------------------
// SimCLR NT-Xent loss, fused.  B=4096, D=128, N=8192, T=0.5, eps=1e-8.
// loss = (1/N) * sum_i [ 2 + ln( sum_{j != i} exp(2*sim_ij - 2) ) - 2*sim_{i, i^B} ]
// sim via v_wmma_f32_16x16x32_f16 on f16 unit vectors (f32 accum).
// B tiles staged in LDS with double-buffered global_load_async_to_lds_b128.
// ---------------------------------------------------------------------------

typedef _Float16 half4_t  __attribute__((ext_vector_type(4)));
typedef _Float16 half8_t  __attribute__((ext_vector_type(8)));
typedef _Float16 half16_t __attribute__((ext_vector_type(16)));
typedef float    v8f      __attribute__((ext_vector_type(8)));

#define D_DIM 128
#define TWO_LOG2E 2.8853900817779268f   // 2*log2(e): exp(2s-2) = exp2((s-1)*2*log2e)
#define COLCHUNKS 16
#define LDS_STRIDE_H 136                // 128 halfs + 8 pad (272 B row stride)
#define LDS_BUF_H (32 * LDS_STRIDE_H)   // one 32-col tile

// ---------------------------------------------------------------------------
// Kernel 0: zero accumulators
// ---------------------------------------------------------------------------
__global__ void zero_kernel(float* __restrict__ p, int n) {
    int i = blockIdx.x * blockDim.x + threadIdx.x;
    if (i < n) p[i] = 0.0f;
}

// ---------------------------------------------------------------------------
// Kernel 1: normalize rows of concat(proj_i, proj_j); write f32 + f16 z.
// ---------------------------------------------------------------------------
__global__ __launch_bounds__(256) void normalize_kernel(
    const float* __restrict__ pi, const float* __restrict__ pj,
    float* __restrict__ zf, _Float16* __restrict__ zh, int Bhalf, int N)
{
    int row  = blockIdx.x * (blockDim.x >> 5) + (threadIdx.x >> 5);
    if (row >= N) return;
    int lane = threadIdx.x & 31;

    const float* src = (row < Bhalf) ? (pi + (size_t)row * D_DIM)
                                     : (pj + (size_t)(row - Bhalf) * D_DIM);
    float4 v = ((const float4*)src)[lane];
    float ss = v.x * v.x + v.y * v.y + v.z * v.z + v.w * v.w;
#pragma unroll
    for (int m = 1; m <= 16; m <<= 1) ss += __shfl_xor(ss, m, 32);

    float s = 1.0f / fmaxf(sqrtf(ss), 1e-8f);
    float4 o;
    o.x = v.x * s; o.y = v.y * s; o.z = v.z * s; o.w = v.w * s;

    ((float4*)(zf + (size_t)row * D_DIM))[lane] = o;
    half4_t h = { (_Float16)o.x, (_Float16)o.y, (_Float16)o.z, (_Float16)o.w };
    ((half4_t*)(zh + (size_t)row * D_DIM))[lane] = h;
}

// ---------------------------------------------------------------------------
// A fragment from global (row-major f16 z).
// A order used here (contiguous k, matched by B): lanes 0-15: M=lane, K=k0..k0+15;
//                     lanes 16-31: M=lane-16, K=k0+16..k0+31.
// ---------------------------------------------------------------------------
__device__ __forceinline__ half16_t load_a_frag(const _Float16* __restrict__ zh,
                                                int rBase, int k0, int lane) {
    int row   = rBase + (lane & 15);
    int khalf = (lane >= 16) ? 16 : 0;
    const _Float16* p = zh + (size_t)row * D_DIM + k0 + khalf;
    half8_t lo = *(const half8_t*)(p);
    half8_t hi = *(const half8_t*)(p + 8);
    return __builtin_shufflevector(lo, hi, 0,1,2,3,4,5,6,7,8,9,10,11,12,13,14,15);
}

// C epilogue: VGPR v, lanes 0-15: row=rowBase+v, lanes 16-31: row=rowBase+v+8;
// col = colBase + (lane&15).  Accumulate exp(2s-2), skip diagonal exactly.
__device__ __forceinline__ void tile_epilogue(const v8f& c, int rowBase, int colBase,
                                              int lane, float* __restrict__ acc) {
    int col  = colBase + (lane & 15);
    int rOff = (lane >= 16) ? 8 : 0;
#pragma unroll
    for (int v = 0; v < 8; ++v) {
        int   row = rowBase + v + rOff;
        float e   = exp2f((c[v] - 1.0f) * TWO_LOG2E);   // v_exp_f32 (TRANS)
        acc[v]   += (row == col) ? 0.0f : e;
    }
}

// ---------------------------------------------------------------------------
// Kernel 2: fused sim-GEMM + row-wise sum of exp(2*sim-2), diagonal excluded.
// Workgroup = 8 waves = 256 rows; B tiles (32 cols x K=128) shared via LDS,
// double-buffered with async global->LDS copies (ASYNCcnt).
// ---------------------------------------------------------------------------
__global__ __launch_bounds__(256) void simclr_gemm_lse(
    const _Float16* __restrict__ zh, float* __restrict__ rowsum, int N)
{
    __shared__ _Float16 smem[2 * LDS_BUF_H];

    const int tid  = threadIdx.x;
    const int lane = tid & 31;
    const int wid  = tid >> 5;

    const int rowBlock = blockIdx.x / COLCHUNKS;
    const int chunk    = blockIdx.x % COLCHUNKS;
    const int rBase    = rowBlock * 256 + wid * 32;
    const int colsPer  = N / COLCHUNKS;
    const int cStart   = chunk * colsPer;
    const int nIters   = colsPer / 32;

    // Per-thread staging addresses: 512 x 16B chunks per tile, 2 per thread.
    const int q0col = tid >> 4,           q0seg = tid & 15;         // chunk tid
    const int q1col = (tid + 256) >> 4,   q1seg = (tid + 256) & 15; // chunk tid+256

    // A fragments (32 rows x K=128) resident in VGPRs.
    half16_t a[8];
#pragma unroll
    for (int kk = 0; kk < 4; ++kk) {
        a[kk]     = load_a_frag(zh, rBase,      kk * 32, lane);
        a[4 + kk] = load_a_frag(zh, rBase + 16, kk * 32, lane);
    }

    float acc[16];
#pragma unroll
    for (int i = 0; i < 16; ++i) acc[i] = 0.0f;

    // ---- async staging of one 32-col tile into LDS buffer `buf` ----
    auto stage_tile = [&](int buf, int cBase) {
        {
            uint64_t ga = (uint64_t)(uintptr_t)(zh + (size_t)(cBase + q0col) * D_DIM + q0seg * 8);
            uint32_t lo = (uint32_t)(uintptr_t)(&smem[buf * LDS_BUF_H + q0col * LDS_STRIDE_H + q0seg * 8]);
            asm volatile("global_load_async_to_lds_b128 %0, %1, off"
                         :: "v"(lo), "v"(ga) : "memory");
        }
        {
            uint64_t ga = (uint64_t)(uintptr_t)(zh + (size_t)(cBase + q1col) * D_DIM + q1seg * 8);
            uint32_t lo = (uint32_t)(uintptr_t)(&smem[buf * LDS_BUF_H + q1col * LDS_STRIDE_H + q1seg * 8]);
            asm volatile("global_load_async_to_lds_b128 %0, %1, off"
                         :: "v"(lo), "v"(ga) : "memory");
        }
    };

    stage_tile(0, cStart);
    asm volatile("s_wait_asynccnt 0x0" ::: "memory");
    __syncthreads();

    int cb = 0;
    for (int it = 0; it < nIters; ++it) {
        int cBase = cStart + it * 32;
        if (it + 1 < nIters) stage_tile(cb ^ 1, cBase + 32);   // overlap with compute

        v8f c00 = {}, c01 = {}, c10 = {}, c11 = {};
        const int bufBase = cb * LDS_BUF_H;
        const int khalf   = (lane >= 16) ? 16 : 0;
        const int colLo   = (lane & 15);
#pragma unroll
        for (int kk = 0; kk < 4; ++kk) {
            int k0 = kk * 32;
            // B order used here (contiguous k, same as A): lanes 0-15: N=lane, K=k0..k0+15; lanes 16-31: K=k0+16..k0+31
            half16_t b0 = *(const half16_t*)(&smem[bufBase + colLo        * LDS_STRIDE_H + k0 + khalf]);
            half16_t b1 = *(const half16_t*)(&smem[bufBase + (16 + colLo) * LDS_STRIDE_H + k0 + khalf]);
            c00 = __builtin_amdgcn_wmma_f32_16x16x32_f16(false, a[kk],     false, b0,
                                                         (short)0, c00, false, false);
            c01 = __builtin_amdgcn_wmma_f32_16x16x32_f16(false, a[kk],     false, b1,
                                                         (short)0, c01, false, false);
            c10 = __builtin_amdgcn_wmma_f32_16x16x32_f16(false, a[4 + kk], false, b0,
                                                         (short)0, c10, false, false);
            c11 = __builtin_amdgcn_wmma_f32_16x16x32_f16(false, a[4 + kk], false, b1,
                                                         (short)0, c11, false, false);
        }
        tile_epilogue(c00, rBase,      cBase,      lane, acc);
        tile_epilogue(c01, rBase,      cBase + 16, lane, acc);
        tile_epilogue(c10, rBase + 16, cBase,      lane, acc + 8);
        tile_epilogue(c11, rBase + 16, cBase + 16, lane, acc + 8);

        asm volatile("s_wait_asynccnt 0x0" ::: "memory");  // next tile landed
        __syncthreads();
        cb ^= 1;
    }

    // Reduce across the 16 lanes striping each row (xor masks 1..8 stay in-half).
#pragma unroll
    for (int m = 1; m <= 8; m <<= 1) {
#pragma unroll
        for (int i = 0; i < 16; ++i) acc[i] += __shfl_xor(acc[i], m, 32);
    }
    if ((lane & 15) == 0) {
        int rOff = (lane >= 16) ? 8 : 0;
#pragma unroll
        for (int v = 0; v < 8; ++v) {
            atomicAdd(&rowsum[rBase + v + rOff],      acc[v]);
            atomicAdd(&rowsum[rBase + 16 + v + rOff], acc[8 + v]);
        }
    }
}

// ---------------------------------------------------------------------------
// Kernel 3: positive-pair dot (exact f32) + per-row loss, block-reduced.
// ---------------------------------------------------------------------------
__global__ __launch_bounds__(256) void pos_finalize_kernel(
    const float* __restrict__ zf, const float* __restrict__ rowsum,
    float* __restrict__ lossAcc, int Bhalf, int N)
{
    __shared__ float sdata[8];
    int wid  = threadIdx.x >> 5;
    int lane = threadIdx.x & 31;
    int row  = blockIdx.x * 8 + wid;

    float contrib = 0.0f;
    if (row < N) {
        int partner = (row < Bhalf) ? (row + Bhalf) : (row - Bhalf);
        float4 aV = ((const float4*)(zf + (size_t)row     * D_DIM))[lane];
        float4 bV = ((const float4*)(zf + (size_t)partner * D_DIM))[lane];
        float d = aV.x * bV.x + aV.y * bV.y + aV.z * bV.z + aV.w * bV.w;
#pragma unroll
        for (int m = 1; m <= 16; m <<= 1) d += __shfl_xor(d, m, 32);
        if (lane == 0)
            contrib = (2.0f + logf(rowsum[row])) - 2.0f * d;  // lse_i - pos_i/T
    }
    if (lane == 0) sdata[wid] = contrib;
    __syncthreads();
    if (threadIdx.x == 0) {
        float s = 0.0f;
#pragma unroll
        for (int i = 0; i < 8; ++i) s += sdata[i];
        atomicAdd(lossAcc, s);
    }
}

__global__ void write_out_kernel(const float* __restrict__ lossAcc,
                                 float* __restrict__ out, int N) {
    out[0] = lossAcc[0] / (float)N;
}

// ---------------------------------------------------------------------------
extern "C" void kernel_launch(void* const* d_in, const int* in_sizes, int n_in,
                              void* d_out, int out_size, void* d_ws, size_t ws_size,
                              hipStream_t stream) {
    const float* pi = (const float*)d_in[0];
    const float* pj = (const float*)d_in[1];
    int Bhalf = in_sizes[0] / D_DIM;   // 4096
    int N     = 2 * Bhalf;             // 8192
    float* out = (float*)d_out;

    // Workspace: zf32[N*128] | zf16[N*128] | rowsum[N] | lossAcc[1]
    char*     ws      = (char*)d_ws;
    float*    zf      = (float*)ws;
    _Float16* zh      = (_Float16*)(ws + (size_t)N * D_DIM * 4);
    float*    rowsum  = (float*)(ws + (size_t)N * D_DIM * 6);
    float*    lossAcc = rowsum + N;

    zero_kernel<<<(N + 1 + 255) / 256, 256, 0, stream>>>(rowsum, N + 1);
    normalize_kernel<<<(N + 7) / 8, 256, 0, stream>>>(pi, pj, zf, zh, Bhalf, N);

    int blocks = (N / 256) * COLCHUNKS;   // 512 workgroups, 4096 waves
    simclr_gemm_lse<<<blocks, 256, 0, stream>>>(zh, rowsum, N);

    pos_finalize_kernel<<<(N + 7) / 8, 256, 0, stream>>>(zf, rowsum, lossAcc, Bhalf, N);
    write_out_kernel<<<1, 1, 0, stream>>>(lossAcc, out, N);
}
